// SpatialUniCrossAttention_9603546874538
// MI455X (gfx1250) — matrix operand order repaired and checked by reading, not compiled, not run on hardware
//
#include <hip/hip_runtime.h>
#include <hip/hip_bf16.h>
#include <stdint.h>

typedef __bf16 bf16_t;
typedef __attribute__((ext_vector_type(16))) __bf16 v16bf;
typedef __attribute__((ext_vector_type(8)))  __bf16 bf16x8;
typedef __attribute__((ext_vector_type(8)))  float  v8f;
typedef __attribute__((ext_vector_type(4)))  unsigned int v4u;
typedef __attribute__((ext_vector_type(8)))  int v8i;
typedef __attribute__((ext_vector_type(4)))  int v4i;

constexpr int Bc = 2, Lc = 4096, Cc = 256, Hc = 8, Dc = 32;
constexpr int Mrows = Bc * Lc;  // 8192

#if defined(__has_builtin)
#if __has_builtin(__builtin_amdgcn_tensor_load_to_lds)
#define HAVE_TDM 1
#else
#define HAVE_TDM 0
#endif
#else
#define HAVE_TDM 0
#endif

__device__ __forceinline__ v16bf pack16(bf16x8 lo, bf16x8 hi) {
  return __builtin_shufflevector(lo, hi, 0, 1, 2, 3, 4, 5, 6, 7, 8, 9, 10, 11, 12, 13, 14, 15);
}

// ---- DPP16 cross-lane reductions over each 16-lane row (no LDS traffic) ----
// Butterfly masks {1,2,7,15}: quad_perm xor1, quad_perm xor2, row_half_mirror,
// row_mirror. Generates the full 16-lane subgroup; never crosses row halves.
template <int CTRL>
__device__ __forceinline__ float dpp_mov(float x) {
  int i = __builtin_bit_cast(int, x);
  int r = __builtin_amdgcn_update_dpp(i, i, CTRL, 0xF, 0xF, true);
  return __builtin_bit_cast(float, r);
}
__device__ __forceinline__ float red16_max(float x) {
  x = fmaxf(x, dpp_mov<0xB1>(x));   // quad_perm [1,0,3,2]  (xor 1)
  x = fmaxf(x, dpp_mov<0x4E>(x));   // quad_perm [2,3,0,1]  (xor 2)
  x = fmaxf(x, dpp_mov<0x141>(x));  // row_half_mirror      (xor 7)
  x = fmaxf(x, dpp_mov<0x140>(x));  // row_mirror           (xor 15)
  return x;
}
__device__ __forceinline__ float red16_sum(float x) {
  x += dpp_mov<0xB1>(x);
  x += dpp_mov<0x4E>(x);
  x += dpp_mov<0x141>(x);
  x += dpp_mov<0x140>(x);
  return x;
}

// ---- TDM: 2D tile load (64 rows x 32 bf16) into LDS, D# per ISA 8.3/8.4 ----
__device__ __forceinline__ void tdm_load_k_tile(uint32_t lds_off, uint64_t gaddr) {
#if HAVE_TDM
  // group0: count=1 | lds_addr | global_addr[56:0] | type=2
  v4u g0 = {1u, lds_off, (uint32_t)gaddr,
            (uint32_t)((gaddr >> 32) & 0x1FFFFFFu) | (2u << 30)};
  // group1: data_size=2B; tensor_dim0=32, tensor_dim1=1M (never OOB);
  // tile_dim0=32, tile_dim1=64; tensor_dim0_stride=32 elems
  v8i g1 = {(int)0x00010000u,        // workgroup_mask=0, data_size=1 (2B)
            (int)(32u << 16),        // tensor_dim0[15:0]=32
            0,                       // tensor_dim0[31:16]=0, tensor_dim1[15:0]=0
            (int)(0x10u | (32u << 16)),  // tensor_dim1[31:16]=16 (1M), tile_dim0=32
            64,                      // tile_dim1=64, tile_dim2=0
            32,                      // tensor_dim0_stride[31:0]=32
            0, 0};
  v4i z4 = {0, 0, 0, 0};
#if __clang_major__ >= 23
  v8i z8 = {0, 0, 0, 0, 0, 0, 0, 0};
  __builtin_amdgcn_tensor_load_to_lds(g0, g1, z4, z4, z8, 0);
#else
  __builtin_amdgcn_tensor_load_to_lds(g0, g1, z4, z4, 0);
#endif
#else
  (void)lds_off; (void)gaddr;
#endif
}

__global__ __launch_bounds__(256) void cvt_f32_bf16(const float* __restrict__ s,
                                                    bf16_t* __restrict__ d, int n) {
  int i = blockIdx.x * blockDim.x + threadIdx.x;
  if (i < n) d[i] = (bf16_t)s[i];
}

// ---------------------------------------------------------------------------
// 64x64-tile bf16 WMMA GEMM, C = A[M,K] * W[K,N], K=256 fixed.
// MODE 0: fp32 store (final out @ Wo)
// MODE 1: per-32-col L2 normalize * temperature[head] -> bf16 [b][h][l][d] (Q)
// MODE 2: N=512; cols<256 normalized K, cols>=256 plain V -> bf16 [b][h][l][d]
// ---------------------------------------------------------------------------
template <int MODE>
__global__ __launch_bounds__(128)
void gemm64(const bf16_t* __restrict__ A, const bf16_t* __restrict__ W, int Nn,
            float* __restrict__ outF, bf16_t* __restrict__ outQ,
            bf16_t* __restrict__ outK, bf16_t* __restrict__ outV,
            const float* __restrict__ temp) {
  __shared__ __align__(16) bf16_t As[64][32];
  __shared__ __align__(16) bf16_t Bs[64][32];  // transposed: [n][k]
  const int tid = threadIdx.x;
  const int wave = tid >> 5, lane = tid & 31;
  const int lx = lane & 15, lh = lane >> 4;
  const int m0 = blockIdx.x * 64;
  const int n0 = blockIdx.y * 64;

  v8f acc[4] = {};
  for (int k0 = 0; k0 < Cc; k0 += 32) {
    {  // A tile 64x32: each thread 32 contiguous bytes
      const int r = tid >> 1, h = (tid & 1) * 16;
      const bf16x8* src = (const bf16x8*)(A + (size_t)(m0 + r) * Cc + k0 + h);
      *(bf16x8*)&As[r][h] = src[0];
      *(bf16x8*)&As[r][h + 8] = src[1];
    }
    {  // B tile 32x64, transposed into Bs[n][k]
      const int k = tid >> 2, nof = (tid & 3) * 16;
      const bf16_t* src = W + (size_t)(k0 + k) * Nn + n0 + nof;
#pragma unroll
      for (int j = 0; j < 16; ++j) Bs[nof + j][k] = src[j];
    }
    __syncthreads();
    v16bf af;
    {  // A frag: lane half lh selects K {0-7,16-23} vs {8-15,24-31}
      const int row = 16 * wave + lx;
      const int kb = lh * 8;
      af = pack16(*(const bf16x8*)&As[row][kb], *(const bf16x8*)&As[row][kb + 16]);
    }
#pragma unroll
    for (int t = 0; t < 4; ++t) {  // B frag: lane half lh selects K {0-7,16-23} vs {8-15,24-31}
      const int n = 16 * t + lx;
      const int kb = lh * 8;
      v16bf bb = pack16(*(const bf16x8*)&Bs[n][kb], *(const bf16x8*)&Bs[n][kb + 16]);
      acc[t] = __builtin_amdgcn_wmma_f32_16x16x32_bf16(false, af, false, bb, (short)0,
                                                       acc[t], false, false);
    }
    __syncthreads();
  }

  if (MODE == 0) {
#pragma unroll
    for (int t = 0; t < 4; ++t) {
      const int col = n0 + 16 * t + lx;
#pragma unroll
      for (int r = 0; r < 8; ++r) {
        const int mrow = m0 + 16 * wave + r + 8 * lh;
        outF[(size_t)mrow * Nn + col] = acc[t][r];
      }
    }
  } else {
    const bool isV = (MODE == 2) && (n0 >= Cc);
#pragma unroll
    for (int c = 0; c < 2; ++c) {  // two 32-col head chunks per 64-wide tile
      float scl[8] = {1, 1, 1, 1, 1, 1, 1, 1};
      if (!isV) {
#pragma unroll
        for (int r = 0; r < 8; ++r) {
          float p = acc[2 * c][r] * acc[2 * c][r] + acc[2 * c + 1][r] * acc[2 * c + 1][r];
          p = red16_sum(p);
          scl[r] = 1.0f / fmaxf(sqrtf(p), 1e-12f);
        }
      }
      const int colbase = n0 + 32 * c;
      const int h = (isV ? (colbase - Cc) : colbase) >> 5;
      const float tm = (MODE == 1) ? temp[h] : 1.0f;  // fold temperature into q
      bf16_t* dst = (MODE == 1) ? outQ : (isV ? outV : outK);
#pragma unroll
      for (int tt = 0; tt < 2; ++tt) {
        const int t = 2 * c + tt;
        const int d = 16 * tt + lx;
#pragma unroll
        for (int r = 0; r < 8; ++r) {
          const int mrow = m0 + 16 * wave + r + 8 * lh;
          const int b = mrow >> 12, l = mrow & (Lc - 1);
          const float s = isV ? 1.0f : (scl[r] * tm);
          dst[(((size_t)(b * Hc + h)) * Lc + l) * Dc + d] = (bf16_t)(acc[t][r] * s);
        }
      }
    }
  }
}

// ---------------------------------------------------------------------------
// Flash attention: grid (L/64, B*H), 128 threads. Wave w owns 16 query rows.
// K tile arrives via TDM (wave 0 issues descriptor, s_wait_tensorcnt + barrier
// publishes); V is cooperatively transposed into LDS so its B-frags are
// contiguous b128 loads. Softmax reductions are DPP16 butterflies.
// ---------------------------------------------------------------------------
__global__ __launch_bounds__(128)
void flash_attn(const bf16_t* __restrict__ Qb, const bf16_t* __restrict__ Kb,
                const bf16_t* __restrict__ Vb, bf16_t* __restrict__ ctx) {
  __shared__ __align__(16) bf16_t Ks[64][32];
  __shared__ __align__(16) bf16_t Vt[32][64];      // transposed V: [d][key]
  __shared__ __align__(16) bf16_t Ps[4][16][64];   // per-wave P scratch
  const int tid = threadIdx.x;
  const int wave = tid >> 5, lane = tid & 31;
  const int lx = lane & 15, lh = lane >> 4;
  const int bh = blockIdx.y;
  const int b = bh >> 3, h = bh & 7;
  const size_t base = (size_t)bh * Lc * Dc;
  const int l0 = blockIdx.x * 64;
  const uint32_t ks_off = (uint32_t)(uintptr_t)(void*)&Ks[0][0];  // LDS byte offset

  v16bf qf;  // Q A-frag, resident for whole kernel
  {
    const bf16_t* qp = Qb + base + (size_t)(l0 + 16 * wave + lx) * Dc;
    const int kb = lh * 8;
    qf = pack16(*(const bf16x8*)(qp + kb), *(const bf16x8*)(qp + kb + 16));
  }
  float mrow[8], lrow[8];
#pragma unroll
  for (int r = 0; r < 8; ++r) { mrow[r] = -3.0e38f; lrow[r] = 0.0f; }
  v8f oacc[2] = {};

  for (int n0 = 0; n0 < Lc; n0 += 64) {
#if HAVE_TDM
    if (wave == 0)
      tdm_load_k_tile(ks_off, (uint64_t)(uintptr_t)(Kb + base + (size_t)n0 * Dc));
#else
    {
      const int r = tid >> 1, hh = (tid & 1) * 16;
      const bf16x8* kp = (const bf16x8*)(Kb + base + (size_t)(n0 + r) * Dc + hh);
      *(bf16x8*)&Ks[r][hh] = kp[0];
      *(bf16x8*)&Ks[r][hh + 8] = kp[1];
    }
#endif
    {  // cooperative V tile load, transposed on the fly
      const int r = tid >> 1, hh = (tid & 1) * 16;
      const bf16_t* vp = Vb + base + (size_t)(n0 + r) * Dc + hh;
#pragma unroll
      for (int j = 0; j < 16; ++j) Vt[hh + j][r] = vp[j];
    }
#if HAVE_TDM
    if (wave == 0) __builtin_amdgcn_s_wait_tensorcnt(0);
#endif
    __syncthreads();

    v8f s[4];
#pragma unroll
    for (int t = 0; t < 4; ++t) {  // S = Q * K^T
      const int n = 16 * t + lx;
      const int kb = lh * 8;
      v16bf bb = pack16(*(const bf16x8*)&Ks[n][kb], *(const bf16x8*)&Ks[n][kb + 16]);
      v8f z = {};
      s[t] = __builtin_amdgcn_wmma_f32_16x16x32_bf16(false, qf, false, bb, (short)0,
                                                     z, false, false);
    }

    v16bf vf[2][2];  // V B-frags loaded before the barrier releases the tile
#pragma unroll
    for (int ch = 0; ch < 2; ++ch)
#pragma unroll
      for (int nt = 0; nt < 2; ++nt) {
        const int d = 16 * nt + lx;
        const int kb = 32 * ch + lh * 8;
        vf[ch][nt] = pack16(*(const bf16x8*)&Vt[d][kb], *(const bf16x8*)&Vt[d][kb + 16]);
      }

    // online softmax over this wave's 16 rows (rows live per-VGPR, 16-lane halves)
#pragma unroll
    for (int r = 0; r < 8; ++r) {
      float m = fmaxf(fmaxf(s[0][r], s[1][r]), fmaxf(s[2][r], s[3][r]));
      m = red16_max(m);
      const float mnew = fmaxf(mrow[r], m);
      float rs = 0.0f;
#pragma unroll
      for (int t = 0; t < 4; ++t) {
        const float p = __expf(s[t][r] - mnew);
        s[t][r] = p;
        rs += p;
      }
      rs = red16_sum(rs);
      const float alpha = __expf(mrow[r] - mnew);
      lrow[r] = lrow[r] * alpha + rs;
      mrow[r] = mnew;
      oacc[0][r] *= alpha;
      oacc[1][r] *= alpha;
    }

    // P: D-layout -> per-wave LDS (row-major 16x64)
#pragma unroll
    for (int t = 0; t < 4; ++t)
#pragma unroll
      for (int r = 0; r < 8; ++r)
        Ps[wave][r + 8 * lh][16 * t + lx] = (bf16_t)s[t][r];

    __syncthreads();  // all K/V tile reads done; next iter may overwrite

    v16bf pf[2];  // P A-frags (intra-wave LDS dep; compiler inserts s_wait_dscnt)
#pragma unroll
    for (int ch = 0; ch < 2; ++ch) {
      const int kb = 32 * ch + 8 * lh;
      pf[ch] = pack16(*(const bf16x8*)&Ps[wave][lx][kb],
                      *(const bf16x8*)&Ps[wave][lx][kb + 16]);
    }
#pragma unroll
    for (int ch = 0; ch < 2; ++ch)
#pragma unroll
      for (int nt = 0; nt < 2; ++nt)
        oacc[nt] = __builtin_amdgcn_wmma_f32_16x16x32_bf16(false, pf[ch], false,
                                                           vf[ch][nt], (short)0,
                                                           oacc[nt], false, false);
  }

#pragma unroll
  for (int nt = 0; nt < 2; ++nt) {
    const int d = 16 * nt + lx;
#pragma unroll
    for (int r = 0; r < 8; ++r) {
      const int l = l0 + 16 * wave + r + 8 * lh;
      const float v = oacc[nt][r] / lrow[r];
      ctx[((size_t)(b * Lc + l)) * Cc + h * Dc + d] = (bf16_t)v;
    }
  }
}

extern "C" void kernel_launch(void* const* d_in, const int* in_sizes, int n_in,
                              void* d_out, int out_size, void* d_ws, size_t ws_size,
                              hipStream_t stream) {
  (void)in_sizes; (void)n_in; (void)out_size; (void)ws_size;
  const float* x = (const float*)d_in[0];
  const float* y = (const float*)d_in[1];
  const float* Wq = (const float*)d_in[2];
  const float* Wkv = (const float*)d_in[3];
  const float* Wo = (const float*)d_in[4];
  const float* tp = (const float*)d_in[5];
  float* out = (float*)d_out;

  // workspace layout (bf16 elems): ~24.5 MiB total
  bf16_t* ws = (bf16_t*)d_ws;
  const size_t nXY = (size_t)Bc * Lc * Cc;  // 2097152 (also == B*H*L*d)
  bf16_t* xbf = ws;
  bf16_t* ybf = xbf + nXY;
  bf16_t* wqb = ybf + nXY;
  bf16_t* wkvb = wqb + (size_t)Cc * Cc;
  bf16_t* wob = wkvb + (size_t)Cc * 2 * Cc;
  bf16_t* qb = wob + (size_t)Cc * Cc;
  bf16_t* kb = qb + nXY;
  bf16_t* vb = kb + nXY;
  bf16_t* cx = vb + nXY;

  cvt_f32_bf16<<<(int)(nXY / 256), 256, 0, stream>>>(x, xbf, (int)nXY);
  cvt_f32_bf16<<<(int)(nXY / 256), 256, 0, stream>>>(y, ybf, (int)nXY);
  cvt_f32_bf16<<<Cc * Cc / 256, 256, 0, stream>>>(Wq, wqb, Cc * Cc);
  cvt_f32_bf16<<<Cc * 2 * Cc / 256, 256, 0, stream>>>(Wkv, wkvb, Cc * 2 * Cc);
  cvt_f32_bf16<<<Cc * Cc / 256, 256, 0, stream>>>(Wo, wob, Cc * Cc);

  // Q = norm(x@Wq)*temp ; K,V from y@Wkv (K normalized)
  gemm64<1><<<dim3(Mrows / 64, Cc / 64), 128, 0, stream>>>(
      xbf, wqb, Cc, nullptr, qb, nullptr, nullptr, tp);
  gemm64<2><<<dim3(Mrows / 64, 2 * Cc / 64), 128, 0, stream>>>(
      ybf, wkvb, 2 * Cc, nullptr, nullptr, kb, vb, tp);

  flash_attn<<<dim3(Lc / 64, Bc * Hc), 128, 0, stream>>>(qb, kb, vb, cx);

  // out = ctx @ Wo (fp32)
  gemm64<0><<<dim3(Mrows / 64, Cc / 64), 128, 0, stream>>>(
      cx, wob, Cc, out, nullptr, nullptr, nullptr, tp);
}
